// MultiHeadedAttention_50431505990010
// MI455X (gfx1250) — matrix operand order repaired and checked by reading, not compiled, not run on hardware
//
#include <hip/hip_runtime.h>
#include <hip/hip_bf16.h>

// MI455X / gfx1250: wave32, WMMA 16x16x32 bf16 -> f32 accumulate.
typedef __attribute__((ext_vector_type(16))) __bf16 v16bf;
typedef __attribute__((ext_vector_type(8)))  __bf16 v8bf;
typedef __attribute__((ext_vector_type(4)))  __bf16 v4bf;
typedef __attribute__((ext_vector_type(8)))  float  v8f;
typedef __attribute__((ext_vector_type(4)))  unsigned int u32x4;
typedef __attribute__((ext_vector_type(8)))  int i32x8;
typedef __attribute__((ext_vector_type(4)))  int i32x4;

#define WMMA_BF16(a, b, c) \
  __builtin_amdgcn_wmma_f32_16x16x32_bf16(false, (a), false, (b), (short)0, (c), false, false)

#define HH   16
#define DM   1024
#define DK   64
#define BB   4
#define SS   2048
#define ROWS (BB * SS)   // 8192

// Tile strides (bf16 elems). 40 -> 80B rows: 16B aligned, and 16 lanes doing
// b128 loads at one-row stride start on banks {0,4,8,...,60} -> conflict-free.
#define LDA 40
#define LDB 72   // 144B rows for 64-wide K tiles (= 128B row + 16B TDM pad)

#if defined(__has_builtin)
#if __has_builtin(__builtin_amdgcn_tensor_load_to_lds)
#define HAVE_TDM 1
#endif
#endif

#ifdef HAVE_TDM
// Tensor Data Mover: 2D bf16 tile global->LDS (ISA ch.8 D# layout).
// Optional LDS padding: (pad_amount+1) DWORDs after every 2<<pad_interval DWORDs.
// 6-arg builtin (clang-23 / therock-10.0): (g0, g1, g2, g3, g_extra, cpol).
__device__ __forceinline__ void tdm_load_2d(
    unsigned lds_addr, unsigned long long gaddr,
    unsigned tile_d0, unsigned tile_d1,
    unsigned tensor_d0, unsigned tensor_d1, unsigned stride0,
    bool pad, unsigned pad_interval, unsigned pad_amount) {
  u32x4 g0;
  g0[0] = 1u;                                                      // count=1
  g0[1] = lds_addr;                                                // LDS bytes
  g0[2] = (unsigned)gaddr;                                         // ga[31:0]
  g0[3] = (unsigned)((gaddr >> 32) & 0x01FFFFFFull) | (2u << 30);  // ga[56:32], type=2
  i32x8 g1;
  unsigned d0 = (1u << 16);                                        // data_size=2B
  if (pad) d0 |= (1u << 20) | (pad_interval << 22) | (pad_amount << 25);
  g1[0] = (int)d0;
  g1[1] = (int)(tensor_d0 << 16);                   // tensor_dim0 @ bits 79:48
  g1[2] = (int)((tensor_d0 >> 16) | (tensor_d1 << 16));
  g1[3] = (int)((tensor_d1 >> 16) | (tile_d0 << 16));  // tile_dim0 @ 127:112
  g1[4] = (int)(tile_d1 & 0xFFFFu);                 // tile_dim1; tile_dim2=0
  g1[5] = (int)stride0;                             // tensor_dim0_stride
  g1[6] = 0;
  g1[7] = 0;
  i32x4 z4 = {0, 0, 0, 0};                          // groups 2/3 unused (2D)
  i32x8 z8 = {0, 0, 0, 0, 0, 0, 0, 0};
  __builtin_amdgcn_tensor_load_to_lds(g0, g1, z4, z4, z8, 0);
}
#endif

// Fragment layouts (ISA 7.12.2), wave32, lane = (lm = lane&15, half = lane>>4):
//  A 16x32: lane row m=lm, elements = two contiguous runs k=[half*8,+8) and
//           k=[16+half*8,+8)  -> 2x b128 loads.
//  B 32x16: lane col n=lm, elements = same runs as A, k=[half*8,+8) and k=[16+half*8,+8)
//           from a [n][k]-layout tile -> 2x b128 loads.
//  C/D f32: element r -> m = half*8 + r, n = lm.
union BFrag { v16bf v; v8bf h[2]; };

__device__ __forceinline__ v16bf load_frag(const __bf16* p0, const __bf16* p1) {
  BFrag f;
  f.h[0] = *(const v8bf*)p0;
  f.h[1] = *(const v8bf*)p1;
  return f.v;
}

__device__ __forceinline__ v4bf cvt4(const float4 f) {
  v4bf r = { (__bf16)f.x, (__bf16)f.y, (__bf16)f.z, (__bf16)f.w };
  return r;
}

// ---------------------------------------------------------------------------
// Kernel 1: per-head QKV projection.  X[8192,1024] x W[h][1024,64] + b -> bf16
// grid = (64 rowTiles, 16 heads, 3 {q,k,v}); block = 256 (8 waves x 16 rows).
// Q is pre-scaled by 1/sqrt(d_k)=0.125 so scores come out already scaled.
// ---------------------------------------------------------------------------
__global__ __launch_bounds__(256) void qkv_proj(
    const float* __restrict__ q_in, const float* __restrict__ k_in,
    const float* __restrict__ v_in,
    const float* __restrict__ Wq, const float* __restrict__ Wk,
    const float* __restrict__ Wv,
    const float* __restrict__ bq, const float* __restrict__ bk,
    const float* __restrict__ bv,
    __bf16* __restrict__ Qs, __bf16* __restrict__ Ks, __bf16* __restrict__ Vs) {
  const int z = blockIdx.z;
  const float* X    = (z == 0) ? q_in : (z == 1) ? k_in : v_in;
  const float* W    = (z == 0) ? Wq   : (z == 1) ? Wk   : Wv;
  const float* bias = (z == 0) ? bq   : (z == 1) ? bk   : bv;
  __bf16* dst       = (z == 0) ? Qs   : (z == 1) ? Ks   : Vs;
  const float oscale = (z == 0) ? 0.125f : 1.0f;

  const int h = blockIdx.y;
  const int rowBase = blockIdx.x * 128;

  __shared__ alignas(16) __bf16 Xs[128][LDA];  // A tile [row][k]
  __shared__ alignas(16) __bf16 Wt[64][LDA];   // B tile transposed [n][k]

  const int tid = threadIdx.x, wave = tid >> 5, lane = tid & 31;
  const int lm = lane & 15, half = lane >> 4;

  v8f acc[4] = {};   // 16 rows x 64 cols per wave

  for (int mb = 0; mb < DM; mb += 32) {
    // stage X tile: 128x32 fp32 -> bf16, float4-vectorized, coalesced
    for (int e = tid; e < 128 * 8; e += 256) {
      int r = e >> 3, c = (e & 7) * 4;
      const float4 f = *(const float4*)&X[(size_t)(rowBase + r) * DM + mb + c];
      *(v4bf*)&Xs[r][c] = cvt4(f);
    }
    // stage W tile transposed: Wt[n][kk] = W[h][mb+kk][n] (reads coalesced in n)
    for (int e = tid; e < 32 * 64; e += 256) {
      int kk = e >> 6, n = e & 63;
      Wt[n][kk] = (__bf16)W[(size_t)h * (DM * DK) + (size_t)(mb + kk) * DK + n];
    }
    __syncthreads();

    const v16bf a =
        load_frag(&Xs[wave * 16 + lm][half * 8], &Xs[wave * 16 + lm][16 + half * 8]);
#pragma unroll
    for (int nt = 0; nt < 4; ++nt) {
      const v16bf b =
          load_frag(&Wt[nt * 16 + lm][half * 8], &Wt[nt * 16 + lm][16 + half * 8]);
      acc[nt] = WMMA_BF16(a, b, acc[nt]);
    }
    __syncthreads();
  }

#pragma unroll
  for (int nt = 0; nt < 4; ++nt) {
#pragma unroll
    for (int r = 0; r < 8; ++r) {
      int n   = nt * 16 + lm;
      int row = rowBase + wave * 16 + half * 8 + r;
      int bb  = row >> 11, s = row & (SS - 1);
      float v = (acc[nt][r] + bias[h * DK + n]) * oscale;
      dst[(((size_t)bb * HH + h) * SS + s) * DK + n] = (__bf16)v;
    }
  }
}

// ---------------------------------------------------------------------------
// Kernel 2: column softmax stats (softmax over the QUERY axis).
// grid = (32 col-strips of 64, 64 bh); per column s: m_s = max_q scores,
// r_s = 1/sum_q exp(score - m_s).  Online max/sum per lane, merged via LDS.
// K strip is DMA'd by the Tensor Data Mover ([s][d] is natively B-layout;
// TDM pad inserts 16B per 128B row -> conflict-free 72-elem LDS stride).
// ---------------------------------------------------------------------------
__global__ __launch_bounds__(256) void colstats(
    const __bf16* __restrict__ Qs, const __bf16* __restrict__ Ks,
    float* __restrict__ stats) {
  const int bh = blockIdx.y;
  const int s0 = blockIdx.x * 64;

  __shared__ alignas(16) __bf16 Kl[64][LDB];   // [s][d]
  __shared__ float  Ms[64][16];
  __shared__ float  Ls[64][16];

  const int tid = threadIdx.x, wave = tid >> 5, lane = tid & 31;
  const int lm = lane & 15, half = lane >> 4;

  const __bf16* Kb = Ks + (size_t)bh * SS * DK;
  const __bf16* Qb = Qs + (size_t)bh * SS * DK;

#ifdef HAVE_TDM
  if (wave == 0) {
    tdm_load_2d((unsigned)(size_t)&Kl[0][0],
                (unsigned long long)(size_t)&Kb[(size_t)s0 * DK],
                /*tile*/ DK, 64, /*tensor*/ DK, SS, /*stride0*/ DK,
                /*pad*/ true, /*interval 32dw*/ 4, /*amount 4dw*/ 3);
    __builtin_amdgcn_s_wait_tensorcnt(0);
  }
#else
  for (int e = tid; e < 64 * 8; e += 256) {       // b128 copy, coalesced
    int r = e >> 3, c = (e & 7) * 8;
    *(v8bf*)&Kl[r][c] = *(const v8bf*)&Kb[(size_t)(s0 + r) * DK + c];
  }
#endif
  __syncthreads();

  float mrun[4], lrun[4];
#pragma unroll
  for (int t = 0; t < 4; ++t) { mrun[t] = -3.0e38f; lrun[t] = 0.0f; }

  for (int qt = wave; qt < SS / 16; qt += 8) {
    const size_t qoff = (size_t)(qt * 16 + lm) * DK;
    v16bf aq[2];
#pragma unroll
    for (int kb = 0; kb < 2; ++kb)   // Q A-fragments straight from global: b128
      aq[kb] = load_frag(&Qb[qoff + kb * 32 + half * 8],
                         &Qb[qoff + kb * 32 + 16 + half * 8]);

#pragma unroll
    for (int st = 0; st < 4; ++st) {
      v8f c = {};
#pragma unroll
      for (int kb = 0; kb < 2; ++kb) {
        const v16bf b = load_frag(&Kl[st * 16 + lm][kb * 32 + half * 8],
                                  &Kl[st * 16 + lm][kb * 32 + 16 + half * 8]);
        c = WMMA_BF16(aq[kb], b, c);
      }
      float tmax = c[0];
#pragma unroll
      for (int r = 1; r < 8; ++r) tmax = fmaxf(tmax, c[r]);
      float nm = fmaxf(mrun[st], tmax);
      float l  = lrun[st] * __expf(mrun[st] - nm);
#pragma unroll
      for (int r = 0; r < 8; ++r) l += __expf(c[r] - nm);
      mrun[st] = nm;
      lrun[st] = l;
    }
  }

#pragma unroll
  for (int st = 0; st < 4; ++st) {
    Ms[st * 16 + lm][wave * 2 + half] = mrun[st];
    Ls[st * 16 + lm][wave * 2 + half] = lrun[st];
  }
  __syncthreads();

  if (tid < 64) {
    float M = Ms[tid][0];
    for (int i = 1; i < 16; ++i) M = fmaxf(M, Ms[tid][i]);
    float L = 0.0f;
    for (int i = 0; i < 16; ++i) L += Ls[tid][i] * __expf(Ms[tid][i] - M);
    size_t idx = ((size_t)bh * SS + s0 + tid) * 2;
    stats[idx]     = M;
    stats[idx + 1] = 1.0f / L;
  }
}

// ---------------------------------------------------------------------------
// Kernel 3: ctx = (exp(scores - m_s) * r_s) @ V, flash-style over s chunks.
// grid = (16 q-strips of 128, 64 bh).  K chunk DMA'd by TDM each iteration
// (B-layout for scores), V staged transposed [d][s] (B-layout for ctx GEMM),
// attn tile written in C-layout / re-read in A-layout through per-wave LDS.
// ---------------------------------------------------------------------------
__global__ __launch_bounds__(256) void attn_ctx(
    const __bf16* __restrict__ Qs, const __bf16* __restrict__ Ks,
    const __bf16* __restrict__ Vs, const float* __restrict__ stats,
    __bf16* __restrict__ Ctx) {
  const int bh = blockIdx.y;
  const int q0 = blockIdx.x * 128;

  __shared__ alignas(16) __bf16 Kl[32][LDB];       // [s][d]
  __shared__ alignas(16) __bf16 Vt[64][LDA];       // [d][s]  (transposed)
  __shared__ alignas(16) __bf16 Al[8][16][LDA];    // per-wave attn tile [q][s]

  const int tid = threadIdx.x, wave = tid >> 5, lane = tid & 31;
  const int lm = lane & 15, half = lane >> 4;

  const __bf16* Kb = Ks + (size_t)bh * SS * DK;
  const __bf16* Vb = Vs + (size_t)bh * SS * DK;
  const __bf16* Qb = Qs + (size_t)bh * SS * DK;
  const float*  cst = stats + (size_t)bh * SS * 2;

  const size_t qoff = (size_t)(q0 + wave * 16 + lm) * DK;
  v16bf aq[2];
#pragma unroll
  for (int kb = 0; kb < 2; ++kb)
    aq[kb] = load_frag(&Qb[qoff + kb * 32 + half * 8],
                       &Qb[qoff + kb * 32 + 16 + half * 8]);

  v8f acc[4] = {};

  for (int sc = 0; sc < SS; sc += 32) {
    {
#ifdef HAVE_TDM
      // K chunk: async DMA global->LDS with row padding (B-layout)
      if (wave == 0) {
        tdm_load_2d((unsigned)(size_t)&Kl[0][0],
                    (unsigned long long)(size_t)&Kb[(size_t)sc * DK],
                    DK, 32, DK, SS, DK, true, 4, 3);
      }
#else
      {
        int r = tid >> 3, c = (tid & 7) * 8;
        *(v8bf*)&Kl[r][c] = *(const v8bf*)&Kb[(size_t)(sc + r) * DK + c];
      }
#endif
      // V chunk: b128 gather + transposed LDS scatter
      int r = tid >> 3, c = (tid & 7) * 8;
      v8bf vv = *(const v8bf*)&Vb[(size_t)(sc + r) * DK + c];
#pragma unroll
      for (int j = 0; j < 8; ++j) Vt[c + j][r] = vv[j];
      // prefetch next chunk into cache (gfx1250 global_prefetch_b8)
      if (sc + 32 < SS) {
        if (tid < 32)
          __builtin_prefetch(&Kb[(size_t)(sc + 32) * DK + tid * 64], 0, 0);
        else if (tid < 64)
          __builtin_prefetch(&Vb[(size_t)(sc + 32) * DK + (tid - 32) * 64], 0, 0);
      }
#ifdef HAVE_TDM
      if (wave == 0) __builtin_amdgcn_s_wait_tensorcnt(0);
#endif
    }
    __syncthreads();

#pragma unroll
    for (int st = 0; st < 2; ++st) {
      v8f c = {};
#pragma unroll
      for (int kb = 0; kb < 2; ++kb) {
        const v16bf b = load_frag(&Kl[st * 16 + lm][kb * 32 + half * 8],
                                  &Kl[st * 16 + lm][kb * 32 + 16 + half * 8]);
        c = WMMA_BF16(aq[kb], b, c);
      }
      const int scol = sc + st * 16 + lm;
      const float m  = cst[(size_t)scol * 2];
      const float rs = cst[(size_t)scol * 2 + 1];
#pragma unroll
      for (int r = 0; r < 8; ++r)
        Al[wave][half * 8 + r][st * 16 + lm] = (__bf16)(__expf(c[r] - m) * rs);
    }

    // intra-wave C-layout -> A-layout remap through LDS (DS ops are in-order)
    const v16bf aa =
        load_frag(&Al[wave][lm][half * 8], &Al[wave][lm][16 + half * 8]);

#pragma unroll
    for (int nt = 0; nt < 4; ++nt) {
      const v16bf b =
          load_frag(&Vt[nt * 16 + lm][half * 8], &Vt[nt * 16 + lm][16 + half * 8]);
      acc[nt] = WMMA_BF16(aa, b, acc[nt]);
    }
    __syncthreads();
  }

  const int hh = bh & 15, bb = bh >> 4;
#pragma unroll
  for (int nt = 0; nt < 4; ++nt)
#pragma unroll
    for (int r = 0; r < 8; ++r) {
      int n  = nt * 16 + lm;
      int qq = q0 + wave * 16 + half * 8 + r;
      Ctx[((size_t)bb * SS + qq) * DM + hh * DK + n] = (__bf16)acc[nt][r];
    }
}

// ---------------------------------------------------------------------------
// Kernel 4: out = ctx_cat @ Wo^T + bo.  [8192x1024] x [1024x1024], fp32 out.
// grid = (64 rowTiles of 128, 16 colTiles of 64).  Wo is [n][m] row-major,
// which IS the B-fragment layout -> straight float4->bf16 copy, no transpose.
// ---------------------------------------------------------------------------
__global__ __launch_bounds__(256) void out_proj(
    const __bf16* __restrict__ Ctx, const float* __restrict__ Wo,
    const float* __restrict__ bo, float* __restrict__ Out) {
  const int rowBase = blockIdx.x * 128;
  const int n0 = blockIdx.y * 64;

  __shared__ alignas(16) __bf16 As[128][LDA];   // [row][m]
  __shared__ alignas(16) __bf16 Wt[64][LDA];    // [n][m]

  const int tid = threadIdx.x, wave = tid >> 5, lane = tid & 31;
  const int lm = lane & 15, half = lane >> 4;

  v8f acc[4] = {};

  for (int mb = 0; mb < DM; mb += 32) {
    for (int e = tid; e < 128 * 4; e += 256) {   // ctx already bf16: b128 copy
      int r = e >> 2, c = (e & 3) * 8;
      *(v8bf*)&As[r][c] = *(const v8bf*)&Ctx[(size_t)(rowBase + r) * DM + mb + c];
    }
    for (int e = tid; e < 64 * 8; e += 256) {    // Wo[n][m]: float4 -> bf16
      int n = e >> 3, c = (e & 7) * 4;
      const float4 f = *(const float4*)&Wo[(size_t)(n0 + n) * DM + mb + c];
      *(v4bf*)&Wt[n][c] = cvt4(f);
    }
    if (mb + 32 < DM && tid < 128)
      __builtin_prefetch(&Ctx[(size_t)(rowBase + tid) * DM + mb + 32], 0, 0);
    __syncthreads();

    const v16bf a =
        load_frag(&As[wave * 16 + lm][half * 8], &As[wave * 16 + lm][16 + half * 8]);
#pragma unroll
    for (int nt = 0; nt < 4; ++nt) {
      const v16bf b =
          load_frag(&Wt[nt * 16 + lm][half * 8], &Wt[nt * 16 + lm][16 + half * 8]);
      acc[nt] = WMMA_BF16(a, b, acc[nt]);
    }
    __syncthreads();
  }

#pragma unroll
  for (int nt = 0; nt < 4; ++nt)
#pragma unroll
    for (int r = 0; r < 8; ++r) {
      int n   = n0 + nt * 16 + lm;
      int row = rowBase + wave * 16 + half * 8 + r;
      Out[(size_t)row * DM + n] = acc[nt][r] + bo[n];
    }
}

// ---------------------------------------------------------------------------
extern "C" void kernel_launch(void* const* d_in, const int* in_sizes, int n_in,
                              void* d_out, int out_size, void* d_ws, size_t ws_size,
                              hipStream_t stream) {
  const float* query = (const float*)d_in[0];
  const float* key_  = (const float*)d_in[1];
  const float* value = (const float*)d_in[2];
  const float* Wq    = (const float*)d_in[3];
  const float* bq    = (const float*)d_in[4];
  const float* Wk    = (const float*)d_in[5];
  const float* bk    = (const float*)d_in[6];
  const float* Wv    = (const float*)d_in[7];
  const float* bv    = (const float*)d_in[8];
  const float* Wo    = (const float*)d_in[9];
  const float* bo    = (const float*)d_in[10];
  float* out = (float*)d_out;

  // workspace carve-up (bf16 intermediates + column-softmax stats), ~65 MB
  const size_t nQKV = (size_t)BB * HH * SS * DK;       // 8,388,608 elems
  __bf16* Qs  = (__bf16*)d_ws;
  __bf16* Ksb = Qs + nQKV;
  __bf16* Vsb = Ksb + nQKV;
  float*  st  = (float*)(Vsb + nQKV);                  // 2 floats per (bh,s)
  __bf16* Ctx = (__bf16*)(st + (size_t)BB * HH * SS * 2);

  qkv_proj<<<dim3(ROWS / 128, HH, 3), 256, 0, stream>>>(
      query, key_, value, Wq, Wk, Wv, bq, bk, bv, Qs, Ksb, Vsb);
  colstats<<<dim3(SS / 64, BB * HH), 256, 0, stream>>>(Qs, Ksb, st);
  attn_ctx<<<dim3(SS / 128, BB * HH), 256, 0, stream>>>(Qs, Ksb, Vsb, st, Ctx);
  out_proj<<<dim3(ROWS / 128, DM / 64), 256, 0, stream>>>(Ctx, Wo, bo, out);
}
